// SAModule_Transformer_59923383714425
// MI455X (gfx1250) — matrix operand order repaired and checked by reading, not compiled, not run on hardware
//
#include <hip/hip_runtime.h>
#include <hip/hip_bf16.h>
#include <math.h>

// ---- static config mirroring the reference ----
#define B_      8
#define N_PER   2048
#define D_POS   6
#define M_PER   1024
#define N_TOT   16384          // B_*N_PER
#define M_TOT   8192           // B_*M_PER
#define KNN     64
#define C_IN    64
#define C_OUT   128
#define H_POS   64
#define H2      256
#define C2      256
#define K1PAD   160            // C_OUT + D_POS = 134 padded to 5*32
#define R2      0.25f
#define NEG_BIG (-1.0e30f)

typedef __attribute__((ext_vector_type(16))) _Float16 v16h;
typedef __attribute__((ext_vector_type(8)))  float    v8f;

// ---------------------------------------------------------------------------
// WMMA fragment helpers (wave32; A/B in this code's contiguous K order, C/D in the ISA layout)
// A (16x32 f16): lane L -> row L&15; element e -> K = e + 16*(L>=16)
// B via transposed buffer BT[n][k]: lane L -> n = L&15, K-base = 16*(L>=16),
//   16 contiguous halves.
// C/D (16x16 f32): vgpr v, lane L -> row = v + 8*(L>=16), col = L&15.
// ---------------------------------------------------------------------------
__device__ __forceinline__ v16h frag_a(const _Float16* base, int stride, int lane) {
  const int r  = lane & 15;
  const int hi = (lane >> 4) * 16;
  const _Float16* p = base + r * stride;
  v16h a;
#pragma unroll
  for (int e = 0; e < 8; ++e) a[e] = p[hi + e];
#pragma unroll
  for (int e = 0; e < 8; ++e) a[8 + e] = p[hi + 8 + e];
  return a;
}

__device__ __forceinline__ v16h frag_b(const _Float16* bt, int stride, int lane) {
  const int n  = lane & 15;
  const int kb = (lane >> 4) * 16;
  const _Float16* p = bt + n * stride + kb;
  v16h b;
#pragma unroll
  for (int e = 0; e < 16; ++e) b[e] = p[e];
  return b;
}

__device__ __forceinline__ v8f wmma_f16(v16h a, v16h b, v8f c) {
  return __builtin_amdgcn_wmma_f32_16x16x32_f16(false, a, false, b,
                                                (short)0, c, false, false);
}

// ---------------------------------------------------------------------------
// Prep: fold attn_nn into per-channel vectors; pos_nn(0) constants.
//   w_qa[i] = sum_o W_dst[i,o]*Wa[o]; w_ka[i] = sum_o W_src[i,o]*Wa[o]
//   delta0  = relu(relu(bp1) @ Wp2 + bp2);  d0a = delta0.Wa + ba
// ---------------------------------------------------------------------------
__global__ void prep_kernel(const float* Wsrc, const float* Wdst,
                            const float* bp1, const float* Wp2, const float* bp2,
                            const float* Wa, const float* ba,
                            float* w_qa, float* w_ka, float* delta0, float* d0a) {
  __shared__ float t1[H_POS];
  const int t = threadIdx.x;
  if (t < C_IN) {
    float sq = 0.f, sk = 0.f;
    for (int o = 0; o < C_OUT; ++o) {
      const float wa = Wa[o];
      sq += Wdst[t * C_OUT + o] * wa;
      sk += Wsrc[t * C_OUT + o] * wa;
    }
    w_qa[t] = sq;
    w_ka[t] = sk;
  }
  if (t < H_POS) t1[t] = fmaxf(bp1[t], 0.f);
  __syncthreads();
  if (t < C_OUT) {
    float s = bp2[t];
    for (int h = 0; h < H_POS; ++h) s += t1[h] * Wp2[h * C_OUT + t];
    delta0[t] = fmaxf(s, 0.f);
  }
  __syncthreads();
  if (t == 0) {
    float s = ba[0];
    for (int c = 0; c < C_OUT; ++c) s += delta0[c] * Wa[c];
    d0a[0] = s;
  }
}

// ---------------------------------------------------------------------------
// Transpose + f16-convert all weights once (B-fragments read N-major).
// ---------------------------------------------------------------------------
__global__ void wtrans_kernel(const float* Wlin, const float* Wp1, const float* Wp2,
                              const float* W1, const float* W2,
                              _Float16* WlinT, _Float16* Wp1T, _Float16* Wp2T,
                              _Float16* W1T, _Float16* W2T) {
  const int stride = blockDim.x * gridDim.x;
  const int tid = blockIdx.x * blockDim.x + threadIdx.x;
  for (int i = tid; i < C_OUT * C_IN; i += stride) {       // WlinT[128][64]
    const int n = i / C_IN, k = i % C_IN;
    WlinT[i] = (_Float16)Wlin[k * C_OUT + n];
  }
  for (int i = tid; i < H_POS * 32; i += stride) {         // Wp1T[64][32], K pad
    const int n = i / 32, k = i % 32;
    Wp1T[i] = (_Float16)((k < D_POS) ? Wp1[k * H_POS + n] : 0.f);
  }
  for (int i = tid; i < C_OUT * H_POS; i += stride) {      // Wp2T[128][64]
    const int n = i / H_POS, k = i % H_POS;
    Wp2T[i] = (_Float16)Wp2[k * C_OUT + n];
  }
  for (int i = tid; i < H2 * K1PAD; i += stride) {         // W1T[256][160], K pad
    const int n = i / K1PAD, k = i % K1PAD;
    W1T[i] = (_Float16)((k < (C_OUT + D_POS)) ? W1[k * H2 + n] : 0.f);
  }
  for (int i = tid; i < C2 * H2; i += stride) {            // W2T[256][256]
    const int n = i / H2, k = i % H2;
    W2T[i] = (_Float16)W2[k * C2 + n];
  }
}

// ---------------------------------------------------------------------------
// Farthest point sampling: one block per cloud, serial argmax scan in LDS.
// ---------------------------------------------------------------------------
__global__ void __launch_bounds__(256) fps_kernel(const float* pos, int* idx) {
  __shared__ float dist[N_PER];
  __shared__ float rv[256];
  __shared__ int   ri[256];
  __shared__ int   selS;
  const int b = blockIdx.x, t = threadIdx.x;
  const float* p = pos + (size_t)b * N_PER * D_POS;
  float p0[D_POS];
#pragma unroll
  for (int d = 0; d < D_POS; ++d) p0[d] = p[d];
  for (int i = t; i < N_PER; i += 256) {
    float s = 0.f;
#pragma unroll
    for (int d = 0; d < D_POS; ++d) { float df = p[i * D_POS + d] - p0[d]; s += df * df; }
    dist[i] = s;
  }
  if (t == 0) idx[b * M_PER] = 0;
  __syncthreads();
  for (int it = 1; it < M_PER; ++it) {
    float bv = -1.f; int bi = 0;
    for (int i = t; i < N_PER; i += 256)
      if (dist[i] > bv) { bv = dist[i]; bi = i; }
    rv[t] = bv; ri[t] = bi;
    __syncthreads();
    for (int s = 128; s > 0; s >>= 1) {
      if (t < s && rv[t + s] > rv[t]) { rv[t] = rv[t + s]; ri[t] = ri[t + s]; }
      __syncthreads();
    }
    if (t == 0) { selS = ri[0]; idx[b * M_PER + it] = selS; }
    __syncthreads();
    const int sel = selS;
    float ps[D_POS];
#pragma unroll
    for (int d = 0; d < D_POS; ++d) ps[d] = p[sel * D_POS + d];
    for (int i = t; i < N_PER; i += 256) {
      float s = 0.f;
#pragma unroll
      for (int d = 0; d < D_POS; ++d) { float df = p[i * D_POS + d] - ps[d]; s += df * df; }
      dist[i] = fminf(dist[i], s);
    }
    __syncthreads();
  }
}

// ---------------------------------------------------------------------------
// Radius / k-nearest-in-ball: one block per query; 64 iterated argmins.
// Replicates top_k slot semantics (self point consumes a slot, then invalid).
// ---------------------------------------------------------------------------
__global__ void __launch_bounds__(256) radius_kernel(const float* pos, const int* idx,
                                                     int* nb) {
  __shared__ float d2[N_PER];
  __shared__ float rv[256];
  __shared__ int   ri[256];
  const int m = blockIdx.x, t = threadIdx.x;
  const int b = m / M_PER;
  const int qloc = idx[b * M_PER + (m % M_PER)];
  const float* p = pos + (size_t)b * N_PER * D_POS;
  float q[D_POS];
#pragma unroll
  for (int d = 0; d < D_POS; ++d) q[d] = p[qloc * D_POS + d];
  for (int i = t; i < N_PER; i += 256) {
    float s = 0.f;
#pragma unroll
    for (int d = 0; d < D_POS; ++d) { float df = p[i * D_POS + d] - q[d]; s += df * df; }
    d2[i] = s;
  }
  __syncthreads();
  for (int k = 0; k < KNN; ++k) {
    float bv = 1.0e29f; int bi = -1;
    for (int i = t; i < N_PER; i += 256)
      if (d2[i] < bv) { bv = d2[i]; bi = i; }
    rv[t] = bv; ri[t] = bi;
    __syncthreads();
    for (int s = 128; s > 0; s >>= 1) {
      if (t < s && rv[t + s] < rv[t]) { rv[t] = rv[t + s]; ri[t] = ri[t + s]; }
      __syncthreads();
    }
    if (t == 0) {
      int g = -1;
      if (ri[0] >= 0 && rv[0] <= R2) {
        g = b * N_PER + ri[0];
        if (g == m) g = -1;            // remove_self_loops
      }
      nb[(size_t)m * KNN + k] = g;
      if (ri[0] >= 0) d2[ri[0]] = 1.0e30f;
    }
    __syncthreads();
  }
}

// ---------------------------------------------------------------------------
// qa/ka scalar projections (attn_nn folded into 64-vectors).
// ---------------------------------------------------------------------------
__global__ void qaka_kernel(const float* x, const float* w_qa, const float* w_ka,
                            float* qa, float* ka) {
  const int n = blockIdx.x * blockDim.x + threadIdx.x;
  if (n >= N_TOT) return;
  const float* xr = x + (size_t)n * C_IN;
  float sq = 0.f, sk = 0.f;
  for (int j = 0; j < C_IN; ++j) {
    const float xv = xr[j];
    sq += xv * w_qa[j];
    sk += xv * w_ka[j];
  }
  qa[n] = sq;
  ka[n] = sk;
}

// ---------------------------------------------------------------------------
// v = x @ W_lin  :  [16384,64] x [64,128]  f16 WMMA, f32 accumulate.
// 256-thread blocks (8 waves): wave w owns col tile w, loops 4 row tiles.
// ---------------------------------------------------------------------------
__global__ void __launch_bounds__(256) vgemm_kernel(const float* x,
                                                    const _Float16* WlinT,
                                                    float* vout) {
  __shared__ _Float16 Xh[64 * C_IN];
  const int blk = blockIdx.x, t = threadIdx.x;
  const int r0 = blk * 64;
  for (int i = t; i < 64 * C_IN; i += 256) {
    const int r = i / C_IN, k = i % C_IN;
    Xh[i] = (_Float16)x[(size_t)(r0 + r) * C_IN + k];
  }
  __syncthreads();
  const int wv = t >> 5, lane = t & 31;
  const int ct = wv;                              // 8 col tiles x16 = 128 cols
  for (int rt = 0; rt < 4; ++rt) {
    v8f acc = {};
#pragma unroll
    for (int kc = 0; kc < 2; ++kc) {
      v16h a  = frag_a(Xh + rt * 16 * C_IN + kc * 32, C_IN, lane);
      v16h bf = frag_b(WlinT + ct * 16 * C_IN + kc * 32, C_IN, lane);
      acc = wmma_f16(a, bf, acc);
    }
    const int col   = ct * 16 + (lane & 15);
    const int rbase = rt * 16 + (lane >> 4) * 8;
#pragma unroll
    for (int vv = 0; vv < 8; ++vv)
      vout[(size_t)(r0 + rbase + vv) * C_OUT + col] = acc[vv];
  }
}

// ---------------------------------------------------------------------------
// Fused per-target edge stage: pos_nn MLP (WMMA) + logits + softmax + aggregate.
// One 128-thread block (4 waves) per target m; Delta stays in LDS (never spilled).
// ---------------------------------------------------------------------------
__global__ void __launch_bounds__(128) edge_kernel(
    const float* pos, const int* nb, const float* qa, const float* ka,
    const float* vfeat, const float* bp1g, const float* bp2g, const float* Wa,
    const float* ba, const _Float16* Wp1T, const _Float16* Wp2T,
    const float* delta0, const float* d0a, float* out) {
  __shared__ _Float16 A1[64 * 32];       // 4 KB  : dpos, K padded to 32
  __shared__ _Float16 Hh[64 * H_POS];    // 8 KB  : hidden (post-relu, f16)
  __shared__ float    Delta[64 * C_OUT]; // 32 KB : pos_nn output (f32)
  __shared__ float    wgt[KNN + 1];
  __shared__ int      nbs[KNN];
  __shared__ float    pm[D_POS];
  const int m = blockIdx.x, t = threadIdx.x;
  if (t < D_POS) pm[t] = pos[(size_t)m * D_POS + t];
  if (t < KNN)   nbs[t] = nb[(size_t)m * KNN + t];
  __syncthreads();
  for (int i = t; i < 64 * 32; i += 128) {
    const int e = i >> 5, k = i & 31;
    const int g = nbs[e];
    const float val = (g >= 0 && k < D_POS) ? (pm[k] - pos[(size_t)g * D_POS + k]) : 0.f;
    A1[i] = (_Float16)val;
  }
  __syncthreads();
  const int wv = t >> 5, lane = t & 31;
  // layer 1: [64,32] x [32,64]  -> 16 tiles, 1 K-chunk
  {
    const int ct = wv;
    for (int rt = 0; rt < 4; ++rt) {
      v8f acc = {};
      v16h a  = frag_a(A1 + rt * 16 * 32, 32, lane);
      v16h bf = frag_b(Wp1T + ct * 16 * 32, 32, lane);
      acc = wmma_f16(a, bf, acc);
      const int col   = ct * 16 + (lane & 15);
      const int rbase = rt * 16 + (lane >> 4) * 8;
#pragma unroll
      for (int vv = 0; vv < 8; ++vv)
        Hh[(rbase + vv) * H_POS + col] = (_Float16)fmaxf(acc[vv] + bp1g[col], 0.f);
    }
  }
  __syncthreads();
  // layer 2: [64,64] x [64,128] -> 32 tiles, 2 K-chunks
  for (int cc = 0; cc < 2; ++cc) {
    const int ct = wv + cc * 4;
    for (int rt = 0; rt < 4; ++rt) {
      v8f acc = {};
#pragma unroll
      for (int kc = 0; kc < 2; ++kc) {
        v16h a  = frag_a(Hh + rt * 16 * H_POS + kc * 32, H_POS, lane);
        v16h bf = frag_b(Wp2T + ct * 16 * H_POS + kc * 32, H_POS, lane);
        acc = wmma_f16(a, bf, acc);
      }
      const int col   = ct * 16 + (lane & 15);
      const int rbase = rt * 16 + (lane >> 4) * 8;
#pragma unroll
      for (int vv = 0; vv < 8; ++vv)
        Delta[(rbase + vv) * C_OUT + col] = fmaxf(acc[vv] + bp2g[col], 0.f);
    }
  }
  __syncthreads();
  // logits
  const float qam = qa[m];
  if (t < KNN) {
    const int g = nbs[t];
    float lg = NEG_BIG;
    if (g >= 0) {
      float s = 0.f;
      for (int c = 0; c < C_OUT; ++c) s += Delta[t * C_OUT + c] * Wa[c];
      lg = qam - ka[g] + s + ba[0];
    }
    wgt[t] = lg;
  }
  if (t == KNN) wgt[KNN] = qam - ka[m] + d0a[0];     // self loop (d0a includes ba)
  __syncthreads();
  if (t == 0) {
    float mx = NEG_BIG;
    for (int i = 0; i <= KNN; ++i) mx = fmaxf(mx, wgt[i]);
    float ssum = 0.f;
    for (int i = 0; i <= KNN; ++i) { const float e = __expf(wgt[i] - mx); wgt[i] = e; ssum += e; }
    const float inv = 1.f / ssum;
    for (int i = 0; i <= KNN; ++i) wgt[i] *= inv;
  }
  __syncthreads();
  if (t < KNN && nbs[t] >= 0)
    __builtin_prefetch(vfeat + (size_t)nbs[t] * C_OUT, 0, 1);
  // aggregate: thread t owns channel t
  {
    const int c = t;                                  // 128 threads == C_OUT
    float s = wgt[KNN] * (vfeat[(size_t)m * C_OUT + c] + delta0[c]);
    for (int e = 0; e < KNN; ++e) {
      const int g = nbs[e];
      if (g >= 0) s += wgt[e] * (vfeat[(size_t)g * C_OUT + c] + Delta[e * C_OUT + c]);
    }
    out[(size_t)m * C_OUT + c] = s;
  }
}

// ---------------------------------------------------------------------------
// Tail nodes (m >= M): only self-loop -> out = v + delta0.
// ---------------------------------------------------------------------------
__global__ void tail_kernel(const float* vfeat, const float* delta0, float* out) {
  const int i = blockIdx.x * blockDim.x + threadIdx.x;
  if (i >= (N_TOT - M_TOT) * C_OUT) return;
  const int n = M_TOT + i / C_OUT, c = i % C_OUT;
  out[(size_t)n * C_OUT + c] = vfeat[(size_t)n * C_OUT + c] + delta0[c];
}

// ---------------------------------------------------------------------------
// nn2: relu([out|pos] @ W1 + b1) @ W2 + b2, with in-block 64-row column max.
// 256-thread blocks over 64-row tiles (tiles never straddle clouds).
// ---------------------------------------------------------------------------
__global__ void __launch_bounds__(256) nn2_kernel(
    const float* outf, const float* pos, const _Float16* W1T, const float* b1,
    const _Float16* W2T, const float* b2, float* partial) {
  __shared__ _Float16 Hh[64 * K1PAD];  // 20 KB
  __shared__ _Float16 Y1[64 * H2];     // 32 KB
  const int blk = blockIdx.x, t = threadIdx.x;
  const int r0 = blk * 64;
  for (int i = t; i < 64 * K1PAD; i += 256) {
    const int r = i / K1PAD, k = i % K1PAD;
    float val;
    if (k < C_OUT)              val = outf[(size_t)(r0 + r) * C_OUT + k];
    else if (k < C_OUT + D_POS) val = pos[(size_t)(r0 + r) * D_POS + (k - C_OUT)];
    else                        val = 0.f;
    Hh[i] = (_Float16)val;
  }
  __syncthreads();
  const int wv = t >> 5, lane = t & 31;
  // layer 1: [64,160] x [160,256] -> 64 tiles, 5 K-chunks
  for (int cc = 0; cc < 2; ++cc) {
    const int ct = wv * 2 + cc;                     // 16 col tiles
    for (int rt = 0; rt < 4; ++rt) {
      v8f acc = {};
#pragma unroll
      for (int kc = 0; kc < 5; ++kc) {
        v16h a  = frag_a(Hh + rt * 16 * K1PAD + kc * 32, K1PAD, lane);
        v16h bf = frag_b(W1T + ct * 16 * K1PAD + kc * 32, K1PAD, lane);
        acc = wmma_f16(a, bf, acc);
      }
      const int col   = ct * 16 + (lane & 15);
      const int rbase = rt * 16 + (lane >> 4) * 8;
#pragma unroll
      for (int vv = 0; vv < 8; ++vv)
        Y1[(rbase + vv) * H2 + col] = (_Float16)fmaxf(acc[vv] + b1[col], 0.f);
    }
  }
  __syncthreads();
  // layer 2: [64,256] x [256,256] -> 64 tiles, 8 K-chunks; fold in 64-row max
  for (int cc = 0; cc < 2; ++cc) {
    const int ct  = wv + cc * 8;                    // disjoint cols per wave
    const int col = ct * 16 + (lane & 15);
    float cmax = NEG_BIG;
    for (int rt = 0; rt < 4; ++rt) {
      v8f acc = {};
#pragma unroll
      for (int kc = 0; kc < 8; ++kc) {
        v16h a  = frag_a(Y1 + rt * 16 * H2 + kc * 32, H2, lane);
        v16h bf = frag_b(W2T + ct * 16 * H2 + kc * 32, H2, lane);
        acc = wmma_f16(a, bf, acc);
      }
#pragma unroll
      for (int vv = 0; vv < 8; ++vv) cmax = fmaxf(cmax, acc[vv] + b2[col]);
    }
    cmax = fmaxf(cmax, __shfl_xor(cmax, 16, 32));   // merge row halves
    if ((lane >> 4) == 0) partial[(size_t)blk * C2 + col] = cmax;
  }
}

// ---------------------------------------------------------------------------
// Final pool over the 32 tile-partials per cloud + constant tuple tail.
// ---------------------------------------------------------------------------
__global__ void pool_kernel(const float* partial, float* d_out) {
  const int i = blockIdx.x * blockDim.x + threadIdx.x;
  if (i < B_ * C2) {
    const int b = i / C2, c = i % C2;
    float mx = NEG_BIG;
    for (int j = 0; j < N_PER / 64; ++j)
      mx = fmaxf(mx, partial[(size_t)(b * (N_PER / 64) + j) * C2 + c]);
    d_out[i] = mx;
  }
  if (i < B_ * D_POS) d_out[B_ * C2 + i] = 0.f;                       // zeros [B,6]
  if (i < B_) ((int*)d_out)[B_ * C2 + B_ * D_POS + i] = i;            // arange int32
}

// ---------------------------------------------------------------------------
extern "C" void kernel_launch(void* const* d_in, const int* in_sizes, int n_in,
                              void* d_out, int out_size, void* d_ws, size_t ws_size,
                              hipStream_t stream) {
  (void)in_sizes; (void)n_in; (void)out_size; (void)ws_size;
  const float* x    = (const float*)d_in[0];
  const float* pos  = (const float*)d_in[1];
  /* d_in[2] = batch (implied by layout, unused) */
  const float* Wsrc = (const float*)d_in[3];
  const float* Wdst = (const float*)d_in[4];
  const float* Wlin = (const float*)d_in[5];
  const float* Wp1  = (const float*)d_in[6];
  const float* bp1  = (const float*)d_in[7];
  const float* Wp2  = (const float*)d_in[8];
  const float* bp2  = (const float*)d_in[9];
  const float* Wa   = (const float*)d_in[10];
  const float* ba   = (const float*)d_in[11];
  const float* W1   = (const float*)d_in[12];
  const float* b1   = (const float*)d_in[13];
  const float* W2   = (const float*)d_in[14];
  const float* b2   = (const float*)d_in[15];

  char* ws = (char*)d_ws;
  size_t off = 0;
  auto alloc = [&](size_t bytes) -> void* {
    void* p = ws + off;
    off = (off + bytes + 255) & ~(size_t)255;
    return p;
  };
  int*      idx     = (int*)     alloc((size_t)B_ * M_PER * sizeof(int));
  int*      nb      = (int*)     alloc((size_t)M_TOT * KNN * sizeof(int));
  float*    qa      = (float*)   alloc((size_t)N_TOT * sizeof(float));
  float*    ka      = (float*)   alloc((size_t)N_TOT * sizeof(float));
  float*    w_qa    = (float*)   alloc(C_IN * sizeof(float));
  float*    w_ka    = (float*)   alloc(C_IN * sizeof(float));
  float*    delta0  = (float*)   alloc(C_OUT * sizeof(float));
  float*    d0a     = (float*)   alloc(sizeof(float));
  float*    vbuf    = (float*)   alloc((size_t)N_TOT * C_OUT * sizeof(float));
  float*    outbuf  = (float*)   alloc((size_t)N_TOT * C_OUT * sizeof(float));
  float*    partial = (float*)   alloc((size_t)(N_TOT / 64) * C2 * sizeof(float));
  _Float16* WlinT   = (_Float16*)alloc((size_t)C_OUT * C_IN * sizeof(_Float16));
  _Float16* Wp1T    = (_Float16*)alloc((size_t)H_POS * 32 * sizeof(_Float16));
  _Float16* Wp2T    = (_Float16*)alloc((size_t)C_OUT * H_POS * sizeof(_Float16));
  _Float16* W1T     = (_Float16*)alloc((size_t)H2 * K1PAD * sizeof(_Float16));
  _Float16* W2T     = (_Float16*)alloc((size_t)C2 * H2 * sizeof(_Float16));

  prep_kernel<<<1, 256, 0, stream>>>(Wsrc, Wdst, bp1, Wp2, bp2, Wa, ba,
                                     w_qa, w_ka, delta0, d0a);
  wtrans_kernel<<<128, 256, 0, stream>>>(Wlin, Wp1, Wp2, W1, W2,
                                         WlinT, Wp1T, Wp2T, W1T, W2T);
  fps_kernel<<<B_, 256, 0, stream>>>(pos, idx);
  radius_kernel<<<M_TOT, 256, 0, stream>>>(pos, idx, nb);
  qaka_kernel<<<N_TOT / 128, 128, 0, stream>>>(x, w_qa, w_ka, qa, ka);
  vgemm_kernel<<<N_TOT / 64, 256, 0, stream>>>(x, WlinT, vbuf);
  edge_kernel<<<M_TOT, 128, 0, stream>>>(pos, nb, qa, ka, vbuf, bp1, bp2, Wa, ba,
                                         Wp1T, Wp2T, delta0, d0a, outbuf);
  tail_kernel<<<((N_TOT - M_TOT) * C_OUT + 255) / 256, 256, 0, stream>>>(vbuf, delta0, outbuf);
  nn2_kernel<<<N_TOT / 64, 256, 0, stream>>>(outbuf, pos, W1T, b1, W2T, b2, partial);
  pool_kernel<<<(B_ * C2 + 255) / 256, 256, 0, stream>>>(partial, (float*)d_out);
}
